// DeltaNet_31877247271490
// MI455X (gfx1250) — matrix operand order repaired and checked by reading, not compiled, not run on hardware
//
#include <hip/hip_runtime.h>
#include <hip/hip_bf16.h>
#include <stdint.h>

#define L_SEQ  2048
#define DMODEL 1024
#define NHEADS 4
#define DKH    256
#define NCHUNK 64
#define CSZ    32
#define SLICE  64

typedef __bf16 bf16;
typedef bf16  v16bf __attribute__((ext_vector_type(16)));
typedef float v8f   __attribute__((ext_vector_type(8)));
typedef unsigned int u32x4 __attribute__((ext_vector_type(4)));
typedef unsigned int u32x8 __attribute__((ext_vector_type(8)));

// ---------- bf16 helpers (bit-exact RNE, no reliance on __bf16 arithmetic) ----
__device__ inline bf16 f2bf(float f) {
  union { float f; uint32_t u; } v; v.f = f;
  uint32_t r = v.u + 0x7FFFu + ((v.u >> 16) & 1u);
  union { uint16_t s; bf16 b; } o; o.s = (uint16_t)(r >> 16);
  return o.b;
}

__device__ inline v8f vzero() {
  v8f v;
#pragma unroll
  for (int i = 0; i < 8; ++i) v[i] = 0.f;
  return v;
}

// ---------- CDNA5 async copy: global -> LDS, 16B per lane, ASYNCcnt ----------
__device__ inline void async_copy16(const bf16* gsrc, bf16* ldst) {
  unsigned lds = (unsigned)(uintptr_t)ldst;   // low 32 bits of flat addr = LDS offset
  unsigned long long ga = (unsigned long long)(uintptr_t)gsrc;
  asm volatile("global_load_async_to_lds_b128 %0, %1, off"
               :: "v"(lds), "v"(ga) : "memory");
}
__device__ inline void async_wait0() {
  asm volatile("s_wait_asynccnt 0" ::: "memory");
}

// ---------- CDNA5 TDM: 1-row tensor tile (n bf16 elems) global -> LDS --------
// D# per CDNA5 ISA (count=1, type=2, data_size=2B)
__device__ inline void tdm_load_1d(unsigned lds_off, const bf16* gptr, unsigned n) {
  unsigned long long ga = (unsigned long long)(uintptr_t)gptr;
  u32x4 g0;
  g0[0] = 1u;                                           // count=1, user mode
  g0[1] = lds_off;                                      // lds_addr
  g0[2] = (unsigned)ga;                                 // global_addr[31:0]
  g0[3] = ((unsigned)(ga >> 32) & 0x01FFFFFFu) | (2u << 30); // addr[56:32], type=2
  u32x8 g1;
  g1[0] = 0x00010000u;                                  // data_size=1 (2 bytes)
  g1[1] = (n & 0xFFFFu) << 16;                          // tensor_dim0[15:0]
  g1[2] = (n >> 16) | (1u << 16);                       // tensor_dim0[31:16], tensor_dim1=1
  g1[3] = (n & 0xFFFFu) << 16;                          // tile_dim0 = n
  g1[4] = 0u;                                           // tile_dim1/2 unused
  g1[5] = n;                                            // tensor_dim0_stride
  g1[6] = 0u;
  g1[7] = 0u;
  asm volatile("tensor_load_to_lds %0, %1" :: "s"(g0), "s"(g1) : "memory");
}

// ---------- WMMA operand loaders (CDNA5 16-bit layouts, wave32) ---------------
__device__ inline int kmapA(int e, int hh) { return hh * 16 + e; }

// A[m][k] from row-major src (ld elements per row)
__device__ inline v16bf lda16(const bf16* base, int ld) {
  int lane = threadIdx.x & 31, m = lane & 15, hh = lane >> 4;
  v16bf out;
#pragma unroll
  for (int e = 0; e < 16; ++e) out[e] = base[m * ld + kmapA(e, hh)];
  return out;
}
// A[m][k] = src[k][m]  (transposed source)
__device__ inline v16bf ldaT16(const bf16* base, int ld) {
  int lane = threadIdx.x & 31, m = lane & 15, hh = lane >> 4;
  v16bf out;
#pragma unroll
  for (int e = 0; e < 16; ++e) out[e] = base[kmapA(e, hh) * ld + m];
  return out;
}
// B 32x16 (KxN): lanes 0-15 col n hold K=0..15, lanes 16-31 hold K=16..31
__device__ inline v16bf ldb16(const bf16* base, int ld) {
  int lane = threadIdx.x & 31, n = lane & 15, hh = lane >> 4;
  v16bf out;
#pragma unroll
  for (int e = 0; e < 16; ++e) out[e] = base[(hh * 16 + e) * ld + n];
  return out;
}
// B[k][n] = src[n][k]  (transposed source, e.g. weight rows for x @ W^T)
__device__ inline v16bf ldbT16(const bf16* base, int ld) {
  int lane = threadIdx.x & 31, n = lane & 15, hh = lane >> 4;
  v16bf out;
#pragma unroll
  for (int e = 0; e < 16; ++e) out[e] = base[n * ld + hh * 16 + e];
  return out;
}

__device__ inline v8f wmma_bf16(v16bf a, v16bf b, v8f c) {
  return __builtin_amdgcn_wmma_f32_16x16x32_bf16(false, a, false, b, (short)0, c, false, false);
}

__device__ inline void wmma_storec(float* base, int ld, v8f c) {
  int lane = threadIdx.x & 31, n = lane & 15, h2 = lane >> 4;
#pragma unroll
  for (int r = 0; r < 8; ++r) base[(r + 8 * h2) * ld + n] = c[r];
}
__device__ inline v8f wmma_loadc(const float* base, int ld) {
  int lane = threadIdx.x & 31, n = lane & 15, h2 = lane >> 4;
  v8f c;
#pragma unroll
  for (int r = 0; r < 8; ++r) c[r] = base[(r + 8 * h2) * ld + n];
  return c;
}

// ---------------------------- cast kernel ------------------------------------
__global__ __launch_bounds__(256) void cast_f32_bf16(const float* __restrict__ x,
                                                     bf16* __restrict__ y, int n) {
  int i = blockIdx.x * 256 + threadIdx.x;
  if (i < n) y[i] = f2bf(x[i]);
}

// ---------------------------- WMMA GEMM:  C[M,N] = A[M,K] @ W[N,K]^T ---------
#define BM 64
#define BN 128
#define BK 32
__global__ __launch_bounds__(256) void gemm_bf16_nt(const bf16* __restrict__ A,
                                                    const bf16* __restrict__ W,
                                                    float* __restrict__ C,
                                                    int M, int N, int K) {
  __shared__ __align__(16) bf16 sA[BM][BK];
  __shared__ __align__(16) bf16 sB[BN][BK];
  const int tid = threadIdx.x;
  const int wave = tid >> 5;
  const int bm = blockIdx.y * BM;
  const int bn = blockIdx.x * BN;
  const int mt = wave & 3;   // m-tile 0..3
  const int ng = wave >> 2;  // n-group 0..1 (4 n-tiles each)
  v8f acc[4];
#pragma unroll
  for (int j = 0; j < 4; ++j) acc[j] = vzero();
  const int ar = tid >> 2, ac = (tid & 3) << 3;
  for (int k0 = 0; k0 < K; k0 += BK) {
    // async global -> LDS staging (no VGPR bounce), tracked by ASYNCcnt
    async_copy16(&A[(size_t)(bm + ar) * K + k0 + ac], &sA[ar][ac]);
    async_copy16(&W[(size_t)(bn + ar) * K + k0 + ac], &sB[ar][ac]);
    async_copy16(&W[(size_t)(bn + ar + 64) * K + k0 + ac], &sB[ar + 64][ac]);
    if (k0 + BK < K) {
      __builtin_prefetch(&A[(size_t)(bm + ar) * K + k0 + BK + ac], 0, 3);
      __builtin_prefetch(&W[(size_t)(bn + ar) * K + k0 + BK + ac], 0, 3);
      __builtin_prefetch(&W[(size_t)(bn + ar + 64) * K + k0 + BK + ac], 0, 3);
    }
    async_wait0();
    __syncthreads();
    v16bf a = lda16(&sA[mt * 16][0], BK);
#pragma unroll
    for (int j = 0; j < 4; ++j) {
      v16bf b = ldbT16(&sB[ng * 64 + j * 16][0], BK);
      acc[j] = wmma_bf16(a, b, acc[j]);
    }
    __syncthreads();
  }
  const int lane = tid & 31, n = lane & 15, h2 = lane >> 4;
#pragma unroll
  for (int j = 0; j < 4; ++j) {
    float* dst = C + (size_t)(bm + mt * 16) * N + bn + ng * 64 + j * 16;
#pragma unroll
    for (int r = 0; r < 8; ++r) dst[(size_t)(r + 8 * h2) * N + n] = acc[j][r];
  }
}

// ------------------- depthwise causal conv (K=4) + silu ----------------------
__global__ __launch_bounds__(256) void conv_silu(const float* __restrict__ x,
                                                 const float* __restrict__ w,
                                                 float* __restrict__ y, int L, int D) {
  int idx = blockIdx.x * 256 + threadIdx.x;
  if (idx >= L * D) return;
  int l = idx / D, d = idx % D;
  float acc = 0.f;
#pragma unroll
  for (int j = 0; j < 4; ++j) {
    int ls = l - 3 + j;
    float xv = (ls >= 0) ? x[(size_t)ls * D + d] : 0.f;
    acc += xv * w[d * 4 + j];
  }
  y[idx] = acc / (1.f + __expf(-acc));
}

// ------------------- beta = sigmoid(h @ Wb^T) --------------------------------
__global__ __launch_bounds__(128) void beta_kernel(const float* __restrict__ hs,
                                                   const float* __restrict__ Wb,
                                                   float* __restrict__ beta) {
  int l = blockIdx.x;
  int wave = threadIdx.x >> 5, lane = threadIdx.x & 31;
  float acc = 0.f;
  for (int d = lane; d < DMODEL; d += 32) acc += hs[(size_t)l * DMODEL + d] * Wb[wave * DMODEL + d];
#pragma unroll
  for (int off = 16; off; off >>= 1) acc += __shfl_xor(acc, off, 32);
  if (lane == 0) beta[l * NHEADS + wave] = 1.f / (1.f + __expf(-acc));
}

// --------- per (head, chunk): l2norm, G, T (UT inverse), u, w, attn ----------
__global__ __launch_bounds__(256) void delta_pre(const float* __restrict__ qf,
                                                 const float* __restrict__ kf,
                                                 const float* __restrict__ vf,
                                                 const float* __restrict__ beta,
                                                 bf16* __restrict__ qn_g, bf16* __restrict__ kn_g,
                                                 bf16* __restrict__ w_g, bf16* __restrict__ att_g,
                                                 float* __restrict__ u_g) {
  __shared__ __align__(16) bf16 sQ[CSZ][DKH];
  __shared__ __align__(16) bf16 sK[CSZ][DKH];
  __shared__ __align__(16) bf16 sVB[CSZ][DKH];
  __shared__ __align__(16) bf16 sKB[CSZ][DKH];
  __shared__ float fG[CSZ][CSZ];
  __shared__ float fAtt[CSZ][CSZ];
  __shared__ __align__(16) bf16 sT[CSZ][CSZ];
  const int tid = threadIdx.x, wave = tid >> 5, lane = tid & 31;
  const int h = blockIdx.x / NCHUNK;
  const int ci = blockIdx.x % NCHUNK;
  const int l0 = ci * CSZ;

  // l2norm q,k; v*beta; k*beta (one wave per row, 4 rows each)
#pragma unroll
  for (int i = 0; i < 4; ++i) {
    int r = wave + 8 * i;
    int l = l0 + r;
    const float* qrow = qf + (size_t)l * DMODEL + h * DKH;
    const float* krow = kf + (size_t)l * DMODEL + h * DKH;
    const float* vrow = vf + (size_t)l * DMODEL + h * DKH;
    float bet = beta[l * NHEADS + h];
    float sq = 0.f, sk = 0.f;
    for (int d = lane; d < DKH; d += 32) {
      float x = qrow[d]; sq += x * x;
      float y = krow[d]; sk += y * y;
    }
#pragma unroll
    for (int off = 16; off; off >>= 1) { sq += __shfl_xor(sq, off, 32); sk += __shfl_xor(sk, off, 32); }
    float iq = rsqrtf(sq + 1e-6f), ik = rsqrtf(sk + 1e-6f);
    for (int d = lane; d < DKH; d += 32) {
      float qv = qrow[d] * iq, kv = krow[d] * ik;
      bf16 qb = f2bf(qv), kb = f2bf(kv);
      sQ[r][d] = qb; sK[r][d] = kb;
      sKB[r][d] = f2bf(kv * bet);
      sVB[r][d] = f2bf(vrow[d] * bet);
      qn_g[((size_t)h * L_SEQ + l) * DKH + d] = qb;
      kn_g[((size_t)h * L_SEQ + l) * DKH + d] = kb;
    }
  }
  __syncthreads();

  // waves 0-3: G = (k*beta) @ kn^T ; waves 4-7: Att = qn @ kn^T   (32x32, K=256)
  {
    int mt = (wave >> 1) & 1, nt = wave & 1;
    const bool doAtt = (wave >= 4);
    v8f acc = vzero();
#pragma unroll
    for (int kk = 0; kk < DKH / 32; ++kk) {
      v16bf a = lda16((doAtt ? &sQ[mt * 16][0] : &sKB[mt * 16][0]) + kk * 32, DKH);
      v16bf b = ldbT16(&sK[nt * 16][0] + kk * 32, DKH);
      acc = wmma_bf16(a, b, acc);
    }
    wmma_storec(doAtt ? &fAtt[mt * 16][nt * 16] : &fG[mt * 16][nt * 16], CSZ, acc);
  }
  __syncthreads();

  // masked attn -> global (tril incl diag)
  for (int idx = tid; idx < CSZ * CSZ; idx += 256) {
    int r = idx >> 5, c = idx & 31;
    att_g[(((size_t)h * NCHUNK + ci) * CSZ + r) * CSZ + c] = f2bf(r >= c ? fAtt[r][c] : 0.f);
  }
  // T = (I + strict_lower(G))^{-1} by forward substitution (wave 0, lane = col)
  if (wave == 0) {
    float tcol[CSZ];
    int c = lane;
    for (int r = 0; r < CSZ; ++r) {
      float t = (r == c) ? 1.f : 0.f;
      for (int j = c; j < r; ++j) t -= fG[r][j] * tcol[j];
      tcol[r] = t;
      sT[r][c] = f2bf(t);
    }
  }
  __syncthreads();

  // u = T @ (v*beta), w = T @ (k*beta)   (32x256 each, K=32)
  for (int t = wave; t < 64; t += 8) {
    int isw = t >> 5, tt = t & 31, mt = tt >> 4, nt = tt & 15;
    v16bf a = lda16(&sT[mt * 16][0], CSZ);
    v16bf b = ldb16(isw ? &sKB[0][nt * 16] : &sVB[0][nt * 16], DKH);
    v8f acc = wmma_bf16(a, b, vzero());
    int n = lane & 15, h2 = lane >> 4;
    if (!isw) {
      float* dst = u_g + ((size_t)h * L_SEQ + l0 + mt * 16) * DKH + nt * 16;
#pragma unroll
      for (int r = 0; r < 8; ++r) dst[(size_t)(r + 8 * h2) * DKH + n] = acc[r];
    } else {
      bf16* dst = w_g + ((size_t)h * L_SEQ + l0 + mt * 16) * DKH + nt * 16;
#pragma unroll
      for (int r = 0; r < 8; ++r) dst[(size_t)(r + 8 * h2) * DKH + n] = f2bf(acc[r]);
    }
  }
}

// --------- sequential chunk scan; one block per (head, 64-wide dv slice) -----
__global__ __launch_bounds__(256) void delta_scan(const bf16* __restrict__ qn_g,
                                                  const bf16* __restrict__ kn_g,
                                                  const bf16* __restrict__ w_g,
                                                  const bf16* __restrict__ att_g,
                                                  const float* __restrict__ u_g,
                                                  float* __restrict__ dout) {
  __shared__ float Sf[DKH][SLICE];                 // 64 KB  f32 state slice
  __shared__ __align__(16) bf16 Sb[DKH][SLICE];    // 32 KB  bf16 mirror (WMMA operand)
  __shared__ __align__(16) bf16 sW[CSZ][DKH];
  __shared__ __align__(16) bf16 sQ2[CSZ][DKH];
  __shared__ __align__(16) bf16 sKn[CSZ][DKH];
  __shared__ __align__(16) bf16 sAt[CSZ][CSZ];
  __shared__ float sU[CSZ][SLICE];
  __shared__ __align__(16) bf16 sUa[CSZ][SLICE];
  __shared__ float sO[CSZ][SLICE];
  const int tid = threadIdx.x, wave = tid >> 5, lane = tid & 31;
  const int h = blockIdx.x >> 2;
  const int dv0 = (blockIdx.x & 3) * SLICE;

  for (int i = tid; i < DKH * SLICE; i += 256) {
    (&Sf[0][0])[i] = 0.f;
    (&Sb[0][0])[i] = f2bf(0.f);
  }
  __syncthreads();

  for (int ci = 0; ci < NCHUNK; ++ci) {
    const int l0 = ci * CSZ;
    // TDM: one wave DMAs the three 16KB chunk tiles into LDS (TENSORcnt)
    if (wave == 0) {
      size_t gbase = ((size_t)h * L_SEQ + l0) * DKH;
      tdm_load_1d((unsigned)(uintptr_t)&sW[0][0],  w_g + gbase,  CSZ * DKH);
      tdm_load_1d((unsigned)(uintptr_t)&sQ2[0][0], qn_g + gbase, CSZ * DKH);
      tdm_load_1d((unsigned)(uintptr_t)&sKn[0][0], kn_g + gbase, CSZ * DKH);
      __builtin_amdgcn_s_wait_tensorcnt(0);
    }
    for (int i = tid; i < CSZ * CSZ; i += 256)
      (&sAt[0][0])[i] = att_g[((size_t)h * NCHUNK + ci) * (CSZ * CSZ) + i];
    for (int i = tid; i < CSZ * SLICE; i += 256) {
      int r = i / SLICE, c = i % SLICE;
      sU[r][c] = u_g[((size_t)h * L_SEQ + l0 + r) * DKH + dv0 + c];
    }
    __syncthreads();

    // phase 1: u_adj = u - w@S (tasks 0..7); t1 = q@S (tasks 8..15); K=256
#pragma unroll
    for (int i = 0; i < 2; ++i) {
      int task = wave * 2 + i;
      int tt = task & 7, mt = tt >> 2, nt = tt & 3;
      const bf16* arow = (task < 8) ? &sW[mt * 16][0] : &sQ2[mt * 16][0];
      v8f acc = vzero();
#pragma unroll
      for (int kk = 0; kk < DKH / 32; ++kk) {
        v16bf a = lda16(arow + kk * 32, DKH);
        v16bf b = ldb16(&Sb[kk * 32][nt * 16], SLICE);
        acc = wmma_bf16(a, b, acc);
      }
      int n = lane & 15, h2 = lane >> 4;
      if (task < 8) {
#pragma unroll
        for (int r = 0; r < 8; ++r) {
          int rr = mt * 16 + r + 8 * h2, cc = nt * 16 + n;
          float d2 = sU[rr][cc] - acc[r];
          sU[rr][cc] = d2;
          sUa[rr][cc] = f2bf(d2);
        }
      } else {
        wmma_storec(&sO[mt * 16][nt * 16], SLICE, acc);
      }
    }
    __syncthreads();

    // phase 2: o = t1 + attn@u_adj (tasks 0..7, K=32); S += kn^T @ u_adj (tasks 8..71, K=32)
    for (int task = wave; task < 72; task += 8) {
      if (task < 8) {
        int mt = task >> 2, nt = task & 3;
        v8f acc = wmma_loadc(&sO[mt * 16][nt * 16], SLICE);
        acc = wmma_bf16(lda16(&sAt[mt * 16][0], CSZ), ldb16(&sUa[0][nt * 16], SLICE), acc);
        float* dst = dout + (size_t)(l0 + mt * 16) * DMODEL + h * DKH + dv0 + nt * 16;
        int n = lane & 15, h2 = lane >> 4;
#pragma unroll
        for (int r = 0; r < 8; ++r) dst[(size_t)(r + 8 * h2) * DMODEL + n] = acc[r];
      } else {
        int s = task - 8, mt = s >> 2, nt = s & 3;
        v8f acc = wmma_loadc(&Sf[mt * 16][nt * 16], SLICE);
        acc = wmma_bf16(ldaT16(&sKn[0][mt * 16], DKH), ldb16(&sUa[0][nt * 16], SLICE), acc);
        int n = lane & 15, h2 = lane >> 4;
#pragma unroll
        for (int r = 0; r < 8; ++r) {
          int rr = mt * 16 + r + 8 * h2, cc = nt * 16 + n;
          Sf[rr][cc] = acc[r];
          Sb[rr][cc] = f2bf(acc[r]);
        }
      }
    }
    __syncthreads();
  }
}

// ------------------- fused 3-tap-bank depthwise FIR --------------------------
__global__ __launch_bounds__(256) void fir_kernel(const float* __restrict__ v,
                                                  const float* __restrict__ fs,
                                                  const float* __restrict__ fm,
                                                  const float* __restrict__ fl,
                                                  float* __restrict__ os, float* __restrict__ om,
                                                  float* __restrict__ ol) {
  int idx = blockIdx.x * 256 + threadIdx.x;
  if (idx >= L_SEQ * DMODEL) return;
  int l = idx / DMODEL, c = idx % DMODEL;
  float aS = 0.f, aM = 0.f, aL = 0.f;
  for (int i = 0; i < 63; ++i) {
    int ls = l - i;
    if (ls < 0) break;
    float x = v[(size_t)ls * DMODEL + c];
    aL += x * fl[c * 63 + (62 - i)];
    if (i < 15) aM += x * fm[c * 15 + (14 - i)];
    if (i < 3)  aS += x * fs[c * 3 + (2 - i)];
  }
  os[idx] = aS; om[idx] = aM; ol[idx] = aL;
}

// ----------- gate logits, softmax(temp)+eps floor, 5-way mix, RMSNorm --------
__global__ __launch_bounds__(256) void gate_mix(const float* __restrict__ hs,
                                                const float* __restrict__ gw,
                                                const float* __restrict__ gb,
                                                const float* __restrict__ ltemp,
                                                const float* __restrict__ epsp,
                                                const float* __restrict__ normw,
                                                const float* __restrict__ ls,
                                                const float* __restrict__ lm,
                                                const float* __restrict__ ll,
                                                const float* __restrict__ dd,
                                                const float* __restrict__ vd,
                                                bf16* __restrict__ ob) {
  __shared__ float sH[DMODEL];
  __shared__ float sLog[20];
  __shared__ float sP[NHEADS][5];
  __shared__ float sMix[DMODEL];
  __shared__ float sSum[NHEADS];
  int l = blockIdx.x, tid = threadIdx.x;
  for (int d = tid; d < DMODEL; d += 256) sH[d] = hs[(size_t)l * DMODEL + d];
  __syncthreads();
  int wave = tid >> 5, lane = tid & 31;
  for (int p = wave; p < 20; p += 8) {
    int hh = p / 5, pi = p % 5;
    const float* wrow = gw + (size_t)(hh * 5 + pi) * DMODEL;
    float acc = 0.f;
    for (int d = lane; d < DMODEL; d += 32) acc += sH[d] * wrow[d];
#pragma unroll
    for (int off = 16; off; off >>= 1) acc += __shfl_xor(acc, off, 32);
    if (lane == 0) sLog[p] = acc + gb[hh * 5 + pi];
  }
  __syncthreads();
  if (tid < NHEADS) {
    int hh = tid;
    float t = __expf(ltemp[hh]);
    float mx = -1e30f;
    for (int p = 0; p < 5; ++p) mx = fmaxf(mx, sLog[hh * 5 + p] / t);
    float e[5], se = 0.f;
    for (int p = 0; p < 5; ++p) { e[p] = __expf(sLog[hh * 5 + p] / t - mx); se += e[p]; }
    float eps = fminf(fmaxf(epsp[hh], 0.f), 0.2f);
    for (int p = 0; p < 5; ++p) sP[hh][p] = (e[p] / se) * (1.f - 5.f * eps) + eps;
  }
  __syncthreads();
  for (int c = tid; c < DMODEL; c += 256) {
    int hh = c >> 8;
    size_t g = (size_t)l * DMODEL + c;
    float o = sP[hh][0] * ls[g] + sP[hh][1] * lm[g] + sP[hh][2] * ll[g] +
              sP[hh][3] * dd[g] + sP[hh][4] * vd[g];
    sMix[c] = o;
  }
  __syncthreads();
  if (wave < NHEADS) {
    float acc = 0.f;
    for (int d = lane; d < DKH; d += 32) { float x = sMix[wave * DKH + d]; acc += x * x; }
#pragma unroll
    for (int off = 16; off; off >>= 1) acc += __shfl_xor(acc, off, 32);
    if (lane == 0) sSum[wave] = acc;
  }
  __syncthreads();
  for (int c = tid; c < DMODEL; c += 256) {
    int hh = c >> 8, dv = c & 255;
    float inv = rsqrtf(sSum[hh] / (float)DKH + 1e-5f);
    ob[(size_t)l * DMODEL + c] = f2bf(sMix[c] * inv * normw[dv]);
  }
}

// ------------------------------- launch --------------------------------------
#define MB(x) ((size_t)(x) << 20)

extern "C" void kernel_launch(void* const* d_in, const int* in_sizes, int n_in,
                              void* d_out, int out_size, void* d_ws, size_t ws_size,
                              hipStream_t stream) {
  (void)in_sizes; (void)n_in; (void)out_size; (void)ws_size;
  const float* hs    = (const float*)d_in[0];
  const float* Wq    = (const float*)d_in[1];
  const float* Wk    = (const float*)d_in[2];
  const float* Wv    = (const float*)d_in[3];
  const float* Wb    = (const float*)d_in[4];
  const float* convq = (const float*)d_in[5];
  const float* convk = (const float*)d_in[6];
  const float* convv = (const float*)d_in[7];
  const float* firs  = (const float*)d_in[8];
  const float* firm  = (const float*)d_in[9];
  const float* firl  = (const float*)d_in[10];
  const float* gw    = (const float*)d_in[11];
  const float* gb    = (const float*)d_in[12];
  const float* lt    = (const float*)d_in[13];
  const float* ep    = (const float*)d_in[14];
  const float* nw    = (const float*)d_in[15];
  const float* Wo    = (const float*)d_in[16];

  char* ws = (char*)d_ws;
  bf16*  hb   = (bf16*)(ws + MB(0));    // 4 MiB
  bf16*  wqb  = (bf16*)(ws + MB(4));    // 2
  bf16*  wkb  = (bf16*)(ws + MB(6));    // 2
  bf16*  wvb  = (bf16*)(ws + MB(8));    // 2
  bf16*  wob  = (bf16*)(ws + MB(10));   // 2
  float* pq   = (float*)(ws + MB(12));  // 8
  float* pk   = (float*)(ws + MB(20));  // 8
  float* pv   = (float*)(ws + MB(28));  // 8
  float* qf   = (float*)(ws + MB(36));  // 8
  float* kf   = (float*)(ws + MB(44));  // 8
  float* vf   = (float*)(ws + MB(52));  // 8  (v_direct)
  float* betb = (float*)(ws + MB(60));  // 1
  bf16*  qn   = (bf16*)(ws + MB(61));   // 4
  bf16*  kn   = (bf16*)(ws + MB(65));   // 4
  bf16*  wch  = (bf16*)(ws + MB(69));   // 4
  bf16*  att  = (bf16*)(ws + MB(73));   // 1
  float* ug   = (float*)(ws + MB(74));  // 8
  float* dch  = (float*)(ws + MB(82));  // 8  (delta_out, (L, H*DV))
  float* ofs  = (float*)(ws + MB(90));  // 8
  float* ofm  = (float*)(ws + MB(98));  // 8
  float* ofl  = (float*)(ws + MB(106)); // 8
  bf16*  obm  = (bf16*)(ws + MB(114));  // 4

  const int nHD = L_SEQ * DMODEL;   // 2M
  const int nDD = DMODEL * DMODEL;  // 1M
  cast_f32_bf16<<<nHD / 256, 256, 0, stream>>>(hs, hb, nHD);
  cast_f32_bf16<<<nDD / 256, 256, 0, stream>>>(Wq, wqb, nDD);
  cast_f32_bf16<<<nDD / 256, 256, 0, stream>>>(Wk, wkb, nDD);
  cast_f32_bf16<<<nDD / 256, 256, 0, stream>>>(Wv, wvb, nDD);
  cast_f32_bf16<<<nDD / 256, 256, 0, stream>>>(Wo, wob, nDD);

  dim3 gg(DMODEL / BN, L_SEQ / BM);
  gemm_bf16_nt<<<gg, 256, 0, stream>>>(hb, wqb, pq, L_SEQ, DMODEL, DMODEL);
  gemm_bf16_nt<<<gg, 256, 0, stream>>>(hb, wkb, pk, L_SEQ, DMODEL, DMODEL);
  gemm_bf16_nt<<<gg, 256, 0, stream>>>(hb, wvb, pv, L_SEQ, DMODEL, DMODEL);

  conv_silu<<<nHD / 256, 256, 0, stream>>>(pq, convq, qf, L_SEQ, DMODEL);
  conv_silu<<<nHD / 256, 256, 0, stream>>>(pk, convk, kf, L_SEQ, DMODEL);
  conv_silu<<<nHD / 256, 256, 0, stream>>>(pv, convv, vf, L_SEQ, DMODEL);

  beta_kernel<<<L_SEQ, 128, 0, stream>>>(hs, Wb, betb);

  delta_pre<<<NHEADS * NCHUNK, 256, 0, stream>>>(qf, kf, vf, betb, qn, kn, wch, att, ug);
  delta_scan<<<NHEADS * 4, 256, 0, stream>>>(qn, kn, wch, att, ug, dch);

  fir_kernel<<<nHD / 256, 256, 0, stream>>>(vf, firs, firm, firl, ofs, ofm, ofl);

  gate_mix<<<L_SEQ, 256, 0, stream>>>(hs, gw, gb, lt, ep, nw, ofs, ofm, ofl, dch, vf, obm);

  gemm_bf16_nt<<<gg, 256, 0, stream>>>(obm, wob, (float*)d_out, L_SEQ, DMODEL, DMODEL);
}
